// Plan2VecEncoder_44023414784723
// MI455X (gfx1250) — compile-verified, matrix operand order checked by reading, not run on hardware
//
#include <hip/hip_runtime.h>

// ---------------------------------------------------------------------------
// Plan2Vec encoder for MI455X (gfx1250, wave32).
// Memory-bound GNN: f32 node state, f16 WMMA GEMMs (v_wmma_f32_16x16x32_f16)
// with weights pre-swizzled into WMMA B-fragment layout (coalesced 32B/lane
// loads), 8-entry edge-type bias table, atomic segment sums, wave32 LayerNorm.
// ---------------------------------------------------------------------------

typedef __attribute__((ext_vector_type(16))) _Float16 v16h;
typedef __attribute__((ext_vector_type(8)))  float    v8f;

constexpr int NNODE  = 100000;
constexpr int NEDGE  = 800000;
constexpr int NGRAPH = 512;
constexpr int HID    = 256;
constexpr int CATP   = 352;   // 323 padded to 352 (mult. of 32)
constexpr int OUTD   = 512;

static __host__ __device__ inline int cdiv(int a, int b) { return (a + b - 1) / b; }

// ---- f32 weights -> f16, swizzled into WMMA B-fragment layout --------------
// Fragment for (ntile, ktile) = 32 lanes x 16 halves, contiguous (1KB).
// Lane l holds column n = ntile*16 + (l&15); half h maps to
// k = ktile*32 + ((h&8)<<1) + ((l>>4)<<3) + (h&7)   (ISA 16-bit B layout,
// mirror of the A 16x32 layout). Rows >= `rows` are zero padding.
__global__ __launch_bounds__(256)
void conv_w_swz(const float* __restrict__ W, _Float16* __restrict__ o,
                int rows, int cols, int rowsPad)
{
    int idx = blockIdx.x * 256 + threadIdx.x;
    int tot = rowsPad * cols;
    if (idx >= tot) return;
    int frag   = idx >> 9;            // 512 halves per fragment
    int within = idx & 511;
    int lane   = within >> 4;
    int h      = within & 15;
    int ktiles = rowsPad >> 5;
    int nt = frag / ktiles;
    int kt = frag - nt * ktiles;
    int n  = nt * 16 + (lane & 15);
    int k  = kt * 32 + ((h & 8) << 1) + ((lane >> 4) << 3) + (h & 7);
    o[idx] = (k < rows) ? (_Float16)W[(size_t)k * cols + n] : (_Float16)0.f;
}

__global__ __launch_bounds__(256)
void fillzero(float* __restrict__ p, int n)
{
    int i = blockIdx.x * 256 + threadIdx.x;
    if (i < n) p[i] = 0.f;
}

// ---- h0 = concat(op_embed[op_id], nums), zero-padded to 64 cols -------------
__global__ __launch_bounds__(256)
void init_h(const float* __restrict__ x, const float* __restrict__ opE,
            float* __restrict__ h)
{
    int idx = blockIdx.x * 256 + threadIdx.x;          // NNODE*64 threads
    if (idx >= NNODE * 64) return;
    int i = idx >> 6, c = idx & 63;
    float v;
    if (c < 32)      { int op = (int)x[(size_t)i * 13]; v = opE[op * 32 + c]; }
    else if (c < 44)   v = x[(size_t)i * 13 + 1 + (c - 32)];
    else               v = 0.f;
    h[(size_t)i * HID + c] = v;
}

// ---- bias table: tab[a,c] = edge_embed[a] @ leW + leB  (8 edge types) -------
__global__ __launch_bounds__(256)
void edge_bias(const float* __restrict__ ee, const float* __restrict__ leW,
               const float* __restrict__ leB, float* __restrict__ tab,
               int din, int dpad)
{
    int idx = blockIdx.x * 256 + threadIdx.x;          // 8*dpad threads
    if (idx >= 8 * dpad) return;
    int a = idx / dpad, c = idx - a * dpad;
    float v = 0.f;
    if (c < din) {
        v = leB[c];
#pragma unroll
        for (int j = 0; j < 16; ++j) v += ee[a * 16 + j] * leW[j * din + c];
    }
    tab[a * dpad + c] = v;
}

// ---- z = (1+eps) * h over dpad cols ----------------------------------------
__global__ __launch_bounds__(256)
void scale_copy(const float* __restrict__ h, float* __restrict__ z,
                const float* __restrict__ epsp, int total, int shiftD)
{
    int idx = blockIdx.x * 256 + threadIdx.x;
    if (idx >= total) return;
    int i = idx >> shiftD, c = idx & ((1 << shiftD) - 1);
    z[(size_t)i * HID + c] = (1.f + *epsp) * h[(size_t)i * HID + c];
}

// ---- z[dst] += relu(h[src] + tab[attr]) ------------------------------------
__global__ __launch_bounds__(256)
void edge_scatter(const int* __restrict__ ei, const int* __restrict__ ea,
                  const float* __restrict__ h, const float* __restrict__ tab,
                  float* __restrict__ z, int shift /* log2(dpad/4) */)
{
    int idx = blockIdx.x * 256 + threadIdx.x;          // NEDGE * dpad/4 threads
    int e = idx >> shift;
    if (e >= NEDGE) return;
    int c4 = (idx & ((1 << shift) - 1)) << 2;
    int src = ei[e], dst = ei[NEDGE + e], a = ea[e];
    int dpad = 4 << shift;
    float4 hv = *(const float4*)(h + (size_t)src * HID + c4);
    float4 bv = *(const float4*)(tab + a * dpad + c4);
    float4 m;
    m.x = fmaxf(hv.x + bv.x, 0.f); m.y = fmaxf(hv.y + bv.y, 0.f);
    m.z = fmaxf(hv.z + bv.z, 0.f); m.w = fmaxf(hv.w + bv.w, 0.f);
    float* zp = z + (size_t)dst * HID + c4;
    atomicAdd(zp + 0, m.x); atomicAdd(zp + 1, m.y);
    atomicAdd(zp + 2, m.z); atomicAdd(zp + 3, m.w);
}

// ---- WMMA GEMM: C[M,N] = act(A[M,K](f32,lda) @ Bswz[K,N](f16) + bias[N]) ---
// Block = 256 threads = 8 waves, block tile 32x128; each wave: 16 rows x 32
// cols via two v_wmma_f32_16x16x32_f16, A fragment reused across the N pair.
// B is pre-swizzled: one fragment = 32 lanes x 16 contiguous halves, so each
// lane does one 32B contiguous load per fragment (fully coalesced).
// Requires M%32==0, N%128==0, K%32==0 (launch shapes guarantee it).
template<int ACT>  // 0 none, 1 relu, 2 leaky
__global__ __launch_bounds__(256)
void wmma_gemm(const float* __restrict__ A, const _Float16* __restrict__ B,
               const float* __restrict__ bias, float* __restrict__ C,
               int N, int K, int lda)
{
    const int lane  = threadIdx.x & 31;
    const int wave  = threadIdx.x >> 5;
    const int wm    = wave & 1;
    const int wn    = wave >> 1;
    const int m0    = blockIdx.x * 32 + wm * 16;
    const int n0    = blockIdx.y * 128 + wn * 32;
    const int l16   = lane & 15;
    const int khalf = (lane >> 4) << 3;                // +8 K offset for hi lanes

    const int ktiles = K >> 5;
    const int ntile  = n0 >> 4;
    const _Float16* bp0 = B + ((size_t)ntile * ktiles * 32 + lane) * 16;
    const _Float16* bp1 = bp0 + (size_t)ktiles * 512;

    v8f acc0 = {}; v8f acc1 = {};

    for (int k0 = 0; k0 < K; k0 += 32) {
        // A fragment: lane holds row m0+l16, halves 0..7 -> K=k0+khalf..+7,
        // halves 8..15 -> K=k0+16+khalf..+23 (ISA 16-bit A 16x32 layout).
        const float* ar = A + (size_t)(m0 + l16) * lda + k0 + khalf;
        float4 f0 = *(const float4*)(ar);
        float4 f1 = *(const float4*)(ar + 4);
        float4 f2 = *(const float4*)(ar + 16);
        float4 f3 = *(const float4*)(ar + 20);
        v16h a;
        a[0]=(_Float16)f0.x;  a[1]=(_Float16)f0.y;  a[2]=(_Float16)f0.z;  a[3]=(_Float16)f0.w;
        a[4]=(_Float16)f1.x;  a[5]=(_Float16)f1.y;  a[6]=(_Float16)f1.z;  a[7]=(_Float16)f1.w;
        a[8]=(_Float16)f2.x;  a[9]=(_Float16)f2.y;  a[10]=(_Float16)f2.z; a[11]=(_Float16)f2.w;
        a[12]=(_Float16)f3.x; a[13]=(_Float16)f3.y; a[14]=(_Float16)f3.z; a[15]=(_Float16)f3.w;

        v16h b0 = *(const v16h*)bp0;                   // 32B contiguous per lane
        v16h b1 = *(const v16h*)bp1;
        bp0 += 512; bp1 += 512;

        acc0 = __builtin_amdgcn_wmma_f32_16x16x32_f16(false, a, false, b0,
                                                      (short)0, acc0, false, false);
        acc1 = __builtin_amdgcn_wmma_f32_16x16x32_f16(false, a, false, b1,
                                                      (short)0, acc1, false, false);
    }

    const int moff = (lane >> 4) ? 8 : 0;              // C/D: M = v + (lane>=16)*8
#pragma unroll
    for (int v = 0; v < 8; ++v) {
        int r = m0 + moff + v;
        float* crow = C + (size_t)r * N;
        int c0 = n0 + l16, c1 = c0 + 16;
        float v0 = acc0[v] + bias[c0];
        float v1 = acc1[v] + bias[c1];
        if (ACT == 1)      { v0 = fmaxf(v0, 0.f); v1 = fmaxf(v1, 0.f); }
        else if (ACT == 2) { v0 = v0 > 0.f ? v0 : 0.1f * v0;
                             v1 = v1 > 0.f ? v1 : 0.1f * v1; }
        crow[c0] = v0; crow[c1] = v1;
    }
}

// ---- LayerNorm(row) + optional residual + leaky, wave32 per row ------------
__global__ __launch_bounds__(256)
void ln_res_leaky(const float* __restrict__ z2, float* __restrict__ h, int useRes)
{
    int row  = blockIdx.x * 8 + (threadIdx.x >> 5);
    int lane = threadIdx.x & 31;
    const float* zr = z2 + (size_t)row * HID;
    float vals[8]; float s = 0.f, s2 = 0.f;
#pragma unroll
    for (int i = 0; i < 8; ++i) {
        float v = zr[lane + i * 32];
        vals[i] = v; s += v; s2 += v * v;
    }
#pragma unroll
    for (int off = 16; off > 0; off >>= 1) {
        s  += __shfl_xor(s,  off, 32);
        s2 += __shfl_xor(s2, off, 32);
    }
    float mu  = s * (1.f / HID);
    float var = s2 * (1.f / HID) - mu * mu;
    float inv = rsqrtf(var + 1e-5f);
    float* hr = h + (size_t)row * HID;
#pragma unroll
    for (int i = 0; i < 8; ++i) {
        float o = (vals[i] - mu) * inv;
        if (useRes) o += hr[lane + i * 32];
        hr[lane + i * 32] = o > 0.f ? o : 0.1f * o;
    }
}

// ---- graph pooling ----------------------------------------------------------
__global__ __launch_bounds__(256)
void pool_feat(const float* __restrict__ h, const int* __restrict__ batch,
               float* __restrict__ cat)
{
    int idx = blockIdx.x * 256 + threadIdx.x;          // NNODE*256 threads
    if (idx >= NNODE * HID) return;
    int i = idx >> 8, c = idx & 255;                   // whole block = one node
    int b = batch[i];
    atomicAdd(cat + (size_t)b * CATP + c, h[(size_t)i * HID + c]);
}

__global__ __launch_bounds__(256)
void pool_scal(const float* __restrict__ x, const int* __restrict__ batch,
               float* __restrict__ scal)
{
    int i = blockIdx.x * 256 + threadIdx.x;
    if (i >= NNODE) return;
    int b = batch[i];
    atomicAdd(scal + b,            1.f);
    atomicAdd(scal + NGRAPH + b,   x[(size_t)i * 13 + 5]);   // fanout
    atomicAdd(scal + 2*NGRAPH + b, x[(size_t)i * 13 + 4]);   // cost
}

__global__ __launch_bounds__(256)
void pool_fin(const float* __restrict__ scal, float* __restrict__ cat)
{
    int g = blockIdx.x * 256 + threadIdx.x;
    if (g >= NGRAPH) return;
    float cnt = scal[g];
    float d = fmaxf(cnt, 1.f);
    float* cr = cat + (size_t)g * CATP;
    cr[256] = cnt;
    cr[257] = scal[NGRAPH + g] / d;
    cr[258] = scal[2 * NGRAPH + g] / d;
}

// ---- masked mean of token embeddings ---------------------------------------
__global__ __launch_bounds__(64)
void text_feat(const int* __restrict__ ids, const float* __restrict__ mask,
               const float* __restrict__ tok, float* __restrict__ cat)
{
    int g = blockIdx.x, t = threadIdx.x;               // 64 dims
    float sum = 0.f, msum = 0.f;
    for (int l = 0; l < 64; ++l) {
        int id   = ids[g * 64 + l];
        float mk = mask[g * 64 + l];
        float e  = (id != 0) ? tok[(size_t)id * 64 + t] : 0.f;
        sum  += e * mk;
        msum += mk;
    }
    cat[(size_t)g * CATP + 259 + t] = sum / fmaxf(msum, 1.f);
}

// ---------------------------------------------------------------------------
extern "C" void kernel_launch(void* const* d_in, const int* in_sizes, int n_in,
                              void* d_out, int out_size, void* d_ws, size_t ws_size,
                              hipStream_t stream)
{
    (void)in_sizes; (void)n_in; (void)out_size; (void)ws_size;

    const float* x        = (const float*)d_in[0];
    const float* op_emb   = (const float*)d_in[1];
    const float* edge_emb = (const float*)d_in[2];
    // layers at 3 + 7*l: (eps, leW, leB, w1, b1, w2, b2)
    const float* token_embed = (const float*)d_in[24];
    const float* mw1 = (const float*)d_in[25];
    const float* mb1 = (const float*)d_in[26];
    const float* mw2 = (const float*)d_in[27];
    const float* mb2 = (const float*)d_in[28];
    const int* edge_index = (const int*)d_in[29];
    const int* edge_attr  = (const int*)d_in[30];
    const int* batch      = (const int*)d_in[31];
    const int* sql_ids    = (const int*)d_in[32];
    const float* sql_mask = (const float*)d_in[33];

    // ---- workspace layout (floats) ----
    const size_t NH = (size_t)NNODE * HID;             // 25.6M
    float* h    = (float*)d_ws;
    float* z    = h  + NH;
    float* t1   = z  + NH;
    float* tab  = t1 + NH;                             // 8*256
    float* scal = tab + 2048;                          // 3*512
    float* cat  = scal + 3 * NGRAPH;                   // 512*352
    float* hid  = cat + (size_t)NGRAPH * CATP;         // 512*256
    _Float16* wh = (_Float16*)(hid + (size_t)NGRAPH * HID);

    const size_t OW1[3] = {0, 81920, 212992};          // f16 offsets
    const size_t OW2[3] = {16384, 147456, 278528};
    const size_t OMW1 = 344064, OMW2 = 434176;

    const int dpad[3]  = {64, 256, 256};
    const int din[3]   = {44, 256, 256};
    const int shC[3]   = {6, 8, 8};                    // log2(dpad)
    const int shS[3]   = {4, 6, 6};                    // log2(dpad/4)

    // ---- weights -> f16, WMMA-B-fragment swizzled (zero-padded K) ----
    conv_w_swz<<<cdiv(64 * 256, 256), 256, 0, stream>>>(
        (const float*)d_in[6], wh + OW1[0], 44, 256, 64);
    conv_w_swz<<<cdiv(256 * 256, 256), 256, 0, stream>>>(
        (const float*)d_in[8], wh + OW2[0], 256, 256, 256);
    for (int l = 1; l < 3; ++l) {
        conv_w_swz<<<cdiv(256 * 256, 256), 256, 0, stream>>>(
            (const float*)d_in[3 + 7 * l + 3], wh + OW1[l], 256, 256, 256);
        conv_w_swz<<<cdiv(256 * 256, 256), 256, 0, stream>>>(
            (const float*)d_in[3 + 7 * l + 5], wh + OW2[l], 256, 256, 256);
    }
    conv_w_swz<<<cdiv(CATP * 256, 256), 256, 0, stream>>>(mw1, wh + OMW1, 323, 256, CATP);
    conv_w_swz<<<cdiv(256 * 512, 256), 256, 0, stream>>>(mw2, wh + OMW2, 256, 512, 256);

    // ---- node init ----
    init_h<<<cdiv(NNODE * 64, 256), 256, 0, stream>>>(x, op_emb, h);

    // ---- 3 GNN layers ----
    for (int l = 0; l < 3; ++l) {
        const float* epsp = (const float*)d_in[3 + 7 * l + 0];
        const float* leW  = (const float*)d_in[3 + 7 * l + 1];
        const float* leB  = (const float*)d_in[3 + 7 * l + 2];
        const float* b1   = (const float*)d_in[3 + 7 * l + 4];
        const float* b2   = (const float*)d_in[3 + 7 * l + 6];

        edge_bias<<<cdiv(8 * dpad[l], 256), 256, 0, stream>>>(
            edge_emb, leW, leB, tab, din[l], dpad[l]);
        scale_copy<<<cdiv(NNODE * dpad[l], 256), 256, 0, stream>>>(
            h, z, epsp, NNODE * dpad[l], shC[l]);
        edge_scatter<<<cdiv(NEDGE * (dpad[l] / 4), 256), 256, 0, stream>>>(
            edge_index, edge_attr, h, tab, z, shS[l]);

        dim3 g1(NNODE / 32, 256 / 128);
        wmma_gemm<1><<<g1, 256, 0, stream>>>(z,  wh + OW1[l], b1, t1, 256, dpad[l], HID);
        wmma_gemm<0><<<g1, 256, 0, stream>>>(t1, wh + OW2[l], b2, z,  256, 256,     HID);
        ln_res_leaky<<<NNODE / 8, 256, 0, stream>>>(z, h, l > 0 ? 1 : 0);
    }

    // ---- pooling + text ----
    fillzero<<<cdiv(NGRAPH * CATP, 256), 256, 0, stream>>>(cat, NGRAPH * CATP);
    fillzero<<<cdiv(3 * NGRAPH, 256), 256, 0, stream>>>(scal, 3 * NGRAPH);
    pool_feat<<<cdiv(NNODE * HID, 256), 256, 0, stream>>>(h, batch, cat);
    pool_scal<<<cdiv(NNODE, 256), 256, 0, stream>>>(x, batch, scal);
    pool_fin<<<cdiv(NGRAPH, 256), 256, 0, stream>>>(scal, cat);
    text_feat<<<NGRAPH, 64, 0, stream>>>(sql_ids, sql_mask, token_embed, cat);

    // ---- head MLP ----
    wmma_gemm<2><<<dim3(NGRAPH / 32, 256 / 128), 256, 0, stream>>>(
        cat, wh + OMW1, mb1, hid, 256, CATP, CATP);             // leaky
    wmma_gemm<0><<<dim3(NGRAPH / 32, OUTD / 128), 256, 0, stream>>>(
        hid, wh + OMW2, mb2, (float*)d_out, OUTD, 256, HID);    // none
}
